// Multi_attention1_30700426232278
// MI455X (gfx1250) — matrix operand order repaired and checked by reading, not compiled, not run on hardware
//
#include <hip/hip_runtime.h>
#include <hip/hip_bf16.h>

// ---------------------------------------------------------------------------
// MI455X (gfx1250) multi-head attention, compute-bound -> everything on WMMA.
// bf16 inputs, f32 accumulate, flash-attention (no materialized 4096x4096
// score matrices), all intermediates (36MB) stay resident in the 192MB L2.
// Dense GEMMs: 16x64 per wave (4 WMMA accumulators share one A fragment),
// software double-buffered k-loop (loads overlap the WMMA quartet), unrolled
// x2 so the ping/pong buffers coalesce via renaming instead of v_mov copies.
// ---------------------------------------------------------------------------

typedef __bf16 bf16;
typedef __attribute__((ext_vector_type(16))) __bf16 v16bf;
typedef __attribute__((ext_vector_type(8)))  __bf16 v8bf;
typedef __attribute__((ext_vector_type(8)))  float  v8f;

#define H_   4096
#define D_   1024
#define DH_  256
#define NH_  4

__device__ static inline v8f wmma_bf16(v16bf a, v16bf b, v8f c) {
  // v_wmma_f32_16x16x32_bf16 : D = A(16x32) * B(32x16) + C(16x16 f32)
  return __builtin_amdgcn_wmma_f32_16x16x32_bf16(false, a, false, b,
                                                 (short)0, c, false, false);
}

// A fragment (16x32 bf16, MxK) from row-major src; per ISA 7.12.2:
// lanes 0-15 row M=lane hold K {0..7, 16..23}; lanes 16-31 hold K {8..15, 24..31}
__device__ static inline v16bf load_frag_A(const bf16* base, int stride, int lane) {
  int row  = lane & 15;
  int koff = (lane >> 4) << 3;            // 0 or 8
  const bf16* p = base + row * stride + koff;
  union { v16bf v; v8bf h[2]; } u;
  u.h[0] = *(const v8bf*)(p);             // 16B -> global/ds _load_b128
  u.h[1] = *(const v8bf*)(p + 16);
  return u.v;
}

// B fragment (32x16 bf16, KxN) given B^T row-major (rows = N, contiguous K):
// lanes 0-15 col N=lane hold K {0..7, 16..23}; lanes 16-31 hold K {8..15, 24..31}
__device__ static inline v16bf load_frag_BT(const bf16* baseT, int stride, int lane) {
  int col  = lane & 15;
  int koff = (lane >> 4) << 3;            // 0 or 8
  const bf16* p = baseT + col * stride + koff;
  union { v16bf v; v8bf h[2]; } u;
  u.h[0] = *(const v8bf*)(p);
  u.h[1] = *(const v8bf*)(p + 16);
  return u.v;
}

__device__ static inline float rowmax16(float v) {
  v = fmaxf(v, __shfl_xor(v, 1, 32));
  v = fmaxf(v, __shfl_xor(v, 2, 32));
  v = fmaxf(v, __shfl_xor(v, 4, 32));
  v = fmaxf(v, __shfl_xor(v, 8, 32));
  return v;
}
__device__ static inline float rowsum16(float v) {
  v += __shfl_xor(v, 1, 32);
  v += __shfl_xor(v, 2, 32);
  v += __shfl_xor(v, 4, 32);
  v += __shfl_xor(v, 8, 32);
  return v;
}

// ------------------------- conversion kernels ------------------------------
__global__ void k_cvt(const float* __restrict__ src, bf16* __restrict__ dst, int n) {
  int i = blockIdx.x * blockDim.x + threadIdx.x;
  if (i < n) dst[i] = (bf16)src[i];
}

// WqT[n][j][k] = Wq[n][k][j]   (so B-fragments load contiguously)
__global__ void k_cvt_wqT(const float* __restrict__ Wq, bf16* __restrict__ WqT) {
  int i = blockIdx.x * blockDim.x + threadIdx.x;
  if (i < NH_ * DH_ * D_) {
    int head = i >> 18;            // 256*1024 = 2^18
    int j    = (i >> 10) & (DH_ - 1);
    int k    = i & (D_ - 1);
    WqT[i] = (bf16)Wq[head * (D_ * DH_) + k * DH_ + j];
  }
}

// ------------------------- qkv = x @ Wq[n] ---------------------------------
// one wave -> 16x64 block; double-buffered k-loop (preload k+32 while 4 WMMAs
// of step k execute), unrolled x2. Writes qkv row-major AND transposed.
__global__ __launch_bounds__(128) void k_gemm_qkv(
    const bf16* __restrict__ xb, const bf16* __restrict__ wqT,
    bf16* __restrict__ qkv, bf16* __restrict__ qkvT) {
  int lane = threadIdx.x & 31;
  int wid  = blockIdx.x * 4 + (threadIdx.x >> 5);   // 0..4095
  int head = wid >> 10;                             // 1024 blocks / head
  int rem  = wid & 1023;
  int m0   = (rem >> 2) << 4;                       // 256 query tiles
  int n0   = (rem & 3) << 6;                        // 4 n-blocks of 64

  const bf16* A  = xb + m0 * D_;
  const bf16* BT = wqT + head * (DH_ * D_) + n0 * D_;
  v8f acc[4];
  v8f vz = {};
#pragma unroll
  for (int j = 0; j < 4; j++) acc[j] = vz;

  // prologue: fragments for k = 0
  v16bf a = load_frag_A(A, D_, lane);
  v16bf b[4];
#pragma unroll
  for (int j = 0; j < 4; j++) b[j] = load_frag_BT(BT + j * 16 * D_, D_, lane);

#pragma unroll 2
  for (int k = 32; k < D_; k += 32) {
    v16bf an = load_frag_A(A + k, D_, lane);        // next-step loads issue
    v16bf bn[4];
#pragma unroll
    for (int j = 0; j < 4; j++)
      bn[j] = load_frag_BT(BT + j * 16 * D_ + k, D_, lane);
#pragma unroll
    for (int j = 0; j < 4; j++)                     // ...while WMMAs execute
      acc[j] = wmma_bf16(a, b[j], acc[j]);
    a = an;
#pragma unroll
    for (int j = 0; j < 4; j++) b[j] = bn[j];
  }
#pragma unroll
  for (int j = 0; j < 4; j++)                       // epilogue
    acc[j] = wmma_bf16(a, b[j], acc[j]);

  int col = lane & 15, hi = lane >> 4;
  bf16* q  = qkv  + head * (H_ * DH_);
  bf16* qT = qkvT + head * (DH_ * H_);
#pragma unroll
  for (int j = 0; j < 4; j++) {
    int nj = n0 + j * 16;
    union { v8bf v; bf16 e[8]; } pk;
#pragma unroll
    for (int r = 0; r < 8; r++) {
      int m = hi * 8 + r;
      bf16 val = (bf16)acc[j][r];
      q[(m0 + m) * DH_ + nj + col] = val;           // row-major
      pk.e[r] = val;
    }
    *(v8bf*)(qT + (nj + col) * H_ + m0 + hi * 8) = pk.v;  // transposed, 16B
  }
}

// ------------------------- flash attention ---------------------------------
// one wave -> 16 query rows of one head; loop keys in blocks of 32.
// S = Q*K^T (16 WMMA), online softmax, P via LDS relayout, Z += P*V (16 WMMA).
__global__ __launch_bounds__(128) void k_attn(
    const bf16* __restrict__ qkv, const bf16* __restrict__ qkvT,
    bf16* __restrict__ zb) {
  __shared__ bf16 ldsP[4 * 16 * 32];          // per-wave 16x32 P tile
  int lane = threadIdx.x & 31;
  int w    = threadIdx.x >> 5;
  int wid  = blockIdx.x * 4 + w;
  int head = wid >> 8;
  int q0   = (wid & 255) << 4;

  const bf16* Qh = qkv  + head * (H_ * DH_);  // also K (Q==K==V here)
  const bf16* VT = qkvT + head * (DH_ * H_);
  bf16* plds = ldsP + w * (16 * 32);
  int col = lane & 15, hi = lane >> 4;

  // Q fragments resident: 8 x v16bf = 64 VGPRs
  v16bf qf[8];
#pragma unroll
  for (int i = 0; i < 8; i++)
    qf[i] = load_frag_A(Qh + q0 * DH_ + i * 32, DH_, lane);

  v8f accz[16];
  v8f vz = {};
#pragma unroll
  for (int t = 0; t < 16; t++) accz[t] = vz;
  float mrun[8], lrun[8];
#pragma unroll
  for (int r = 0; r < 8; r++) { mrun[r] = -3.0e38f; lrun[r] = 0.0f; }

  for (int kb = 0; kb < H_; kb += 32) {
    __builtin_prefetch(Qh + (kb + 32) * DH_, 0, 0);   // global_prefetch_b8

    // ---- scores: two 16x16 tiles over 32 keys, K-dim = 256 ----
    v8f s0 = {}, s1 = {};
#pragma unroll
    for (int kd = 0; kd < 8; kd++) {
      v16bf b0 = load_frag_BT(Qh + kb * DH_ + kd * 32, DH_, lane);
      v16bf b1 = load_frag_BT(Qh + (kb + 16) * DH_ + kd * 32, DH_, lane);
      s0 = wmma_bf16(qf[kd], b0, s0);
      s1 = wmma_bf16(qf[kd], b1, s1);
    }

    // ---- online softmax (row = query, spread over 16 lanes) ----
    float p0[8], p1[8];
#pragma unroll
    for (int r = 0; r < 8; r++) {
      float mx    = rowmax16(fmaxf(s0[r], s1[r]));
      float mnew  = fmaxf(mrun[r], mx);
      float scale = __expf(mrun[r] - mnew);
      p0[r] = __expf(s0[r] - mnew);
      p1[r] = __expf(s1[r] - mnew);
      float rs = rowsum16(p0[r] + p1[r]);
      lrun[r] = lrun[r] * scale + rs;
      mrun[r] = mnew;
#pragma unroll
      for (int t = 0; t < 16; t++) accz[t][r] *= scale;
    }

    // ---- D-layout -> A-fragment layout via LDS ----
#pragma unroll
    for (int r = 0; r < 8; r++) {
      int m = hi * 8 + r;
      plds[m * 32 + col]      = (bf16)p0[r];
      plds[m * 32 + 16 + col] = (bf16)p1[r];
    }
    __syncthreads();          // uniform trip count across all 4 waves
    v16bf pf = load_frag_A(plds, 32, lane);

    // ---- Z(16x256) += P(16x32) * V(32x256): 16 WMMAs ----
#pragma unroll
    for (int t = 0; t < 16; t++) {
      v16bf bv = load_frag_BT(VT + (t * 16) * H_ + kb, H_, lane);
      accz[t] = wmma_bf16(pf, bv, accz[t]);
    }
  }

  // ---- normalize and store z (bf16, concat-heads layout [H][NH*DH]) ----
  float linv[8];
#pragma unroll
  for (int r = 0; r < 8; r++) linv[r] = 1.0f / lrun[r];
#pragma unroll
  for (int t = 0; t < 16; t++) {
    int cdh = head * DH_ + t * 16 + col;
#pragma unroll
    for (int r = 0; r < 8; r++) {
      int row = q0 + hi * 8 + r;
      zb[row * (NH_ * DH_) + cdh] = (bf16)(accz[t][r] * linv[r]);
    }
  }
}

// ------------------------- out = z @ lin_w^T + b ---------------------------
// one wave -> 16x64 block; same double-buffered, x2-unrolled k-loop.
__global__ __launch_bounds__(128) void k_gemm_out(
    const bf16* __restrict__ zb, const bf16* __restrict__ lwb,
    const float* __restrict__ bias, float* __restrict__ out) {
  int lane = threadIdx.x & 31;
  int wid  = blockIdx.x * 4 + (threadIdx.x >> 5);   // 0..4095
  int m0   = (wid >> 4) << 4;                       // 256 row tiles
  int n0   = (wid & 15) << 6;                       // 16 n-blocks of 64

  const bf16* A  = zb + m0 * D_;
  const bf16* BT = lwb + n0 * D_;   // B[f][o]=lin_w[o][f] => B^T == lin_w rows
  v8f acc[4];
  v8f vz = {};
#pragma unroll
  for (int j = 0; j < 4; j++) acc[j] = vz;

  v16bf a = load_frag_A(A, D_, lane);
  v16bf b[4];
#pragma unroll
  for (int j = 0; j < 4; j++) b[j] = load_frag_BT(BT + j * 16 * D_, D_, lane);

#pragma unroll 2
  for (int k = 32; k < D_; k += 32) {
    v16bf an = load_frag_A(A + k, D_, lane);
    v16bf bn[4];
#pragma unroll
    for (int j = 0; j < 4; j++)
      bn[j] = load_frag_BT(BT + j * 16 * D_ + k, D_, lane);
#pragma unroll
    for (int j = 0; j < 4; j++)
      acc[j] = wmma_bf16(a, b[j], acc[j]);
    a = an;
#pragma unroll
    for (int j = 0; j < 4; j++) b[j] = bn[j];
  }
#pragma unroll
  for (int j = 0; j < 4; j++)
    acc[j] = wmma_bf16(a, b[j], acc[j]);

  int col = lane & 15, hi = lane >> 4;
#pragma unroll
  for (int j = 0; j < 4; j++) {
    int nj = n0 + j * 16;
    float bv = bias[nj + col];
#pragma unroll
    for (int r = 0; r < 8; r++) {
      int m = hi * 8 + r;
      out[(m0 + m) * D_ + nj + col] = acc[j][r] + bv;
    }
  }
}

// ---------------------------------------------------------------------------
extern "C" void kernel_launch(void* const* d_in, const int* in_sizes, int n_in,
                              void* d_out, int out_size, void* d_ws, size_t ws_size,
                              hipStream_t stream) {
  (void)in_sizes; (void)n_in; (void)out_size; (void)ws_size;
  const float* x   = (const float*)d_in[0];   // [4096,1024]
  const float* Wq  = (const float*)d_in[1];   // [4,1024,256]
  const float* lw  = (const float*)d_in[2];   // [1024,1024]
  const float* lb  = (const float*)d_in[3];   // [1024]
  float* out = (float*)d_out;                 // [4096,1024] f32

  char* ws = (char*)d_ws;                     // 36 MB total, fits in L2
  bf16* xb   = (bf16*)(ws);                          //  8 MB
  bf16* wqT  = (bf16*)(ws + (size_t)(8u  << 20));    //  2 MB
  bf16* lwb  = (bf16*)(ws + (size_t)(10u << 20));    //  2 MB
  bf16* qkv  = (bf16*)(ws + (size_t)(12u << 20));    //  8 MB
  bf16* qkvT = (bf16*)(ws + (size_t)(20u << 20));    //  8 MB
  bf16* zb   = (bf16*)(ws + (size_t)(28u << 20));    //  8 MB

  // 1) precision conversion (memory-bound, ~30MB total: negligible vs compute)
  k_cvt    <<<(H_ * D_) / 256, 256, 0, stream>>>(x, xb, H_ * D_);
  k_cvt_wqT<<<(NH_ * DH_ * D_) / 256, 256, 0, stream>>>(Wq, wqT);
  k_cvt    <<<(D_ * NH_ * DH_) / 256, 256, 0, stream>>>(lw, lwb, D_ * NH_ * DH_);

  // 2) qkv = x @ Wq[n]  (4096 waves, 16x64 block each)
  k_gemm_qkv<<<1024, 128, 0, stream>>>(xb, wqT, qkv, qkvT);

  // 3) flash attention (1024 waves: 4 heads x 256 query tiles)
  k_attn<<<256, 128, 0, stream>>>(qkv, qkvT, zb);

  // 4) output projection + bias (4096 waves, 16x64 block each)
  k_gemm_out<<<1024, 128, 0, stream>>>(zb, lwb, lb, out);
}
